// MainNetworkTrainable_34720515621484
// MI455X (gfx1250) — matrix operand order repaired and checked by reading, not compiled, not run on hardware
//
#include <hip/hip_runtime.h>

// ---------------------------------------------------------------------------
// CDNA5 (gfx1250) HyperFast forward pass, v3.
// All large GEMMs via V_WMMA_F32_16X16X32_BF16 (bf16 A/B, f32 accum).
// v3: 256x128 block tiles (64x64 per wave -> 16 WMMA : 16 ds_load), and
// register-staged prefetch so the s_wait_loadcnt for the next tile lands
// AFTER the current tile's WMMA burst (global latency overlapped).
// ---------------------------------------------------------------------------

#define N_ROWS 4096
#define D_IN   1024
#define RF_DIM 4096
#define PC_DIM 1024
#define NC_CLS 10
#define CLIPV  27.6f

typedef __bf16          v16bf __attribute__((ext_vector_type(16)));
typedef unsigned short  u16x8 __attribute__((ext_vector_type(8)));
typedef float           v8f   __attribute__((ext_vector_type(8)));
typedef unsigned short  ushort_t;

union ABfrag { v16bf v; u16x8 h[2]; };

// fp32 -> bf16 (round half up), scalar.
static __device__ __forceinline__ ushort_t f2bf(float f) {
  unsigned int u = __builtin_bit_cast(unsigned int, f);
  return (ushort_t)((u + 0x8000u) >> 16);
}

// two fp32 -> packed bf16x2 via v_perm_b32 (lo in bits[15:0]).
static __device__ __forceinline__ unsigned int pack2bf(float lo, float hi) {
  unsigned int ul = __builtin_bit_cast(unsigned int, lo) + 0x8000u;
  unsigned int uh = __builtin_bit_cast(unsigned int, hi) + 0x8000u;
  return __builtin_amdgcn_perm(uh, ul, 0x07060302u);
}

static __device__ __forceinline__ float wred(float v) {
#pragma unroll
  for (int o = 16; o > 0; o >>= 1) v += __shfl_xor(v, o, 32);
  return v;
}

// ---------------------------------------------------------------------------
// fp32 -> bf16 flat convert, 8 elems/thread (n must be multiple of 8).
// ---------------------------------------------------------------------------
__global__ __launch_bounds__(256) void cvt_bf16_kernel(
    const float* __restrict__ src, ushort_t* __restrict__ dst, int n) {
  const int i = (blockIdx.x * 256 + threadIdx.x) * 8;
  if (i + 8 > n) return;
  float4 a = *(const float4*)(src + i);
  float4 b = *(const float4*)(src + i + 4);
  uint4 o;
  o.x = pack2bf(a.x, a.y);
  o.y = pack2bf(a.z, a.w);
  o.z = pack2bf(b.x, b.y);
  o.w = pack2bf(b.z, b.w);
  *(uint4*)(dst + i) = o;
}

// fp32 [K x C] -> bf16 transposed [C x K]
__global__ __launch_bounds__(256) void cvtT_bf16_kernel(
    const float* __restrict__ W, ushort_t* __restrict__ Wt, int K, int C) {
  const int t = blockIdx.x * 256 + threadIdx.x;   // output index, k fastest
  if (t >= K * C) return;
  const int k = t % K, c = t / K;
  Wt[t] = f2bf(W[(size_t)k * C + c]);
}

// ---------------------------------------------------------------------------
// C[M x Nc] = epilogue( A[M x K] * B[Nc x K]^T ), bf16 WMMA, f32 accumulate.
// Block tile 256(M) x 128(N) x 32(K); 8 waves, each 64x64 (4x4 WMMA tiles).
// MODE 0: relu(x + bias[col])
// MODE 1: clip(x - bias[col], +-CLIP)            (PCA: bias = mean @ comp^T)
// MODE 2: relu(x + bias[col] + res[row,col])     (residual trunk layer)
// MODE 3: x                                      (Gram matrix)
// WRITEF: store fp32 C;  WRITEB: store bf16 mirror Cb (feeds next GEMM).
// ---------------------------------------------------------------------------
template <int MODE, int WRITEF, int WRITEB>
__global__ __launch_bounds__(256) void gemm_bf16_wmma(
    const ushort_t* __restrict__ Ab, const ushort_t* __restrict__ Bb,
    const float* __restrict__ bias, const float* __restrict__ res,
    float* __restrict__ C, ushort_t* __restrict__ Cb, int M, int Nc, int K) {
  __shared__ alignas(32) ushort_t sA[2][256 * 32];   // 2 x 16 KB
  __shared__ alignas(32) ushort_t sB[2][128 * 32];   // 2 x  8 KB

  const int tid  = threadIdx.x;
  const int lane = tid & 31, wave = tid >> 5;
  const int lm = lane & 15, kh = lane >> 4;       // col-in-tile, half select
  const int bm = blockIdx.y * 256, bn = blockIdx.x * 128;
  const int ldc = Nc;
  const int m0 = (wave >> 1) * 64;                // 4 waves along M
  const int n0 = (wave & 1) * 64;                 // 2 waves along N

  // staging addresses (per thread): A row = tid (4x uint4); B row = tid>>1
  const int brow = tid >> 1;
  const int q    = (tid & 1) * 2;

  v8f acc[4][4];
#pragma unroll
  for (int i = 0; i < 4; ++i)
#pragma unroll
    for (int j = 0; j < 4; ++j)
#pragma unroll
      for (int e = 0; e < 8; ++e) acc[i][j][e] = 0.f;

  const int nk = K >> 5;
  uint4 ra0, ra1, ra2, ra3, rb0, rb1;

  // ---- prologue: load + store tile 0 --------------------------------------
  {
    const uint4* sa = (const uint4*)(Ab + (size_t)(bm + tid) * K);
    ra0 = sa[0]; ra1 = sa[1]; ra2 = sa[2]; ra3 = sa[3];
    const uint4* sb = (const uint4*)(Bb + (size_t)(bn + brow) * K);
    rb0 = sb[q]; rb1 = sb[q + 1];
    uint4* da = (uint4*)(sA[0] + tid * 32);
    da[0] = ra0; da[1] = ra1; da[2] = ra2; da[3] = ra3;
    uint4* db = (uint4*)(sB[0] + brow * 32);
    db[q] = rb0; db[q + 1] = rb1;
  }
  __syncthreads();

  for (int ks = 0; ks < nk; ++ks) {
    const int buf = ks & 1;
    const bool pf = (ks + 1 < nk);
    // ---- issue next tile's global loads (no wait yet) ----------------------
    if (pf) {
      const int k0 = (ks + 1) << 5;
      const uint4* sa = (const uint4*)(Ab + (size_t)(bm + tid) * K + k0);
      ra0 = sa[0]; ra1 = sa[1]; ra2 = sa[2]; ra3 = sa[3];
      const uint4* sb = (const uint4*)(Bb + (size_t)(bn + brow) * K + k0);
      rb0 = sb[q]; rb1 = sb[q + 1];
    }
    // ---- compute on current buffer ----------------------------------------
    ABfrag bfr[4];
#pragma unroll
    for (int nt = 0; nt < 4; ++nt) {
      // B frag (32x16 bf16): lane<16 -> K {0..7,16..23} at N=lane, lane>=16 -> +8
      const ushort_t* pb = &sB[buf][(n0 + nt * 16 + lm) * 32 + kh * 8];
      bfr[nt].h[0] = *(const u16x8*)(pb);
      bfr[nt].h[1] = *(const u16x8*)(pb + 16);
    }
#pragma unroll
    for (int mt = 0; mt < 4; ++mt) {
      // A frag (16x32 bf16): lane<16 -> K {0..7,16..23}, lane>=16 -> +8
      ABfrag af;
      const ushort_t* pa = &sA[buf][(m0 + mt * 16 + lm) * 32 + kh * 8];
      af.h[0] = *(const u16x8*)(pa);
      af.h[1] = *(const u16x8*)(pa + 16);
#pragma unroll
      for (int nt = 0; nt < 4; ++nt)
        acc[mt][nt] = __builtin_amdgcn_wmma_f32_16x16x32_bf16(
            false, af.v, false, bfr[nt].v, (short)0, acc[mt][nt],
            false, false);
    }
    // ---- commit next tile to the other LDS buffer (wait lands here) -------
    if (pf) {
      uint4* da = (uint4*)(sA[buf ^ 1] + tid * 32);
      da[0] = ra0; da[1] = ra1; da[2] = ra2; da[3] = ra3;
      uint4* db = (uint4*)(sB[buf ^ 1] + brow * 32);
      db[q] = rb0; db[q + 1] = rb1;
    }
    __syncthreads();
  }

  // Epilogue. C/D layout: row = 8*kh + v, col = lm within each 16x16 tile.
#pragma unroll
  for (int mt = 0; mt < 4; ++mt) {
#pragma unroll
    for (int nt = 0; nt < 4; ++nt) {
      const int gc  = bn + n0 + nt * 16 + lm;
      const int gr0 = bm + m0 + mt * 16 + kh * 8;
#pragma unroll
      for (int v = 0; v < 8; ++v) {
        const int gr = gr0 + v;
        float x = acc[mt][nt][v];
        if (MODE == 0) { x += bias[gc]; x = fmaxf(x, 0.f); }
        else if (MODE == 1) { x -= bias[gc]; x = fminf(fmaxf(x, -CLIPV), CLIPV); }
        else if (MODE == 2) { x += bias[gc] + res[(size_t)gr * ldc + gc]; x = fmaxf(x, 0.f); }
        if (WRITEF) C[(size_t)gr * ldc + gc] = x;
        if (WRITEB) Cb[(size_t)gr * ldc + gc] = f2bf(x);
      }
    }
  }
}

// mb[p] = dot(vec, mat[p,:])  (one wave per row)
__global__ __launch_bounds__(256) void rowdot_kernel(
    const float* __restrict__ vec, const float* __restrict__ mat,
    float* __restrict__ outv, int rows, int D) {
  const int row = blockIdx.x * 8 + (threadIdx.x >> 5);
  const int lane = threadIdx.x & 31;
  if (row >= rows) return;
  const float* m = mat + (size_t)row * D;
  float s = 0.f;
  for (int d = lane; d < D; d += 32) s += vec[d] * m[d];
  s = wred(s);
  if (lane == 0) outv[row] = s;
}

// sq[i] = ||X[i,:]||^2  (one wave per row)
__global__ __launch_bounds__(256) void sqnorm_kernel(
    const float* __restrict__ X, float* __restrict__ sq, int rows, int D) {
  const int row = blockIdx.x * 8 + (threadIdx.x >> 5);
  const int lane = threadIdx.x & 31;
  if (row >= rows) return;
  const float* x = X + (size_t)row * D;
  float s = 0.f;
  for (int d = lane; d < D; d += 32) { float v = x[d]; s += v * v; }
  s = wred(s);
  if (lane == 0) sq[row] = s;
}

// out[i,c] = pen[i,:] @ W3[:,c] + b3[c]   (10 waves per block, wave per class)
__global__ __launch_bounds__(320) void logits_kernel(
    const float* __restrict__ pen, const float* __restrict__ W3,
    const float* __restrict__ b3, float* __restrict__ out, int D, int C) {
  const int i = blockIdx.x;
  const int c = threadIdx.x >> 5, lane = threadIdx.x & 31;
  const float* p = pen + (size_t)i * D;
  float s = 0.f;
  for (int k = lane; k < D; k += 32) s += p[k] * W3[(size_t)k * C + c];
  s = wred(s);
  if (lane == 0) out[(size_t)i * C + c] = s + b3[c];
}

// LOO 1-NN row argmin over d2 = sq[i]+sq[j]-2*G[i,j], then scatter-add bias
// to the predicted-class logit (first-minimum tie-break like jnp.argmin).
__global__ __launch_bounds__(256) void argmin_bias_kernel(
    const float* __restrict__ G, const float* __restrict__ sq,
    const int* __restrict__ y, const float* __restrict__ bias,
    float* __restrict__ out, int n, int C) {
  __shared__ float sval[256];
  __shared__ int   sidx[256];
  const int i = blockIdx.x, tid = threadIdx.x;
  const float sqi = sq[i];
  const float* g = G + (size_t)i * n;
  float best = 3.4e38f;
  int bidx = 0x7fffffff;
  for (int j = tid; j < n; j += 256) {
    if (j == i) continue;
    float d = sqi + sq[j] - 2.f * g[j];
    if (d < best) { best = d; bidx = j; }
  }
  sval[tid] = best; sidx[tid] = bidx;
  __syncthreads();
  for (int s = 128; s > 0; s >>= 1) {
    if (tid < s) {
      float o = sval[tid + s]; int oi = sidx[tid + s];
      if (o < sval[tid] || (o == sval[tid] && oi < sidx[tid])) {
        sval[tid] = o; sidx[tid] = oi;
      }
    }
    __syncthreads();
  }
  if (tid == 0) {
    const int nbr = sidx[0];
    out[(size_t)i * C + y[nbr]] += bias[0];
  }
}

// ---------------------------------------------------------------------------
extern "C" void kernel_launch(void* const* d_in, const int* in_sizes, int n_in,
                              void* d_out, int out_size, void* d_ws,
                              size_t ws_size, hipStream_t stream) {
  (void)in_sizes; (void)n_in; (void)out_size; (void)ws_size;
  const float* X        = (const float*)d_in[0];
  const int*   y        = (const int*)  d_in[1];
  const float* rf_w     = (const float*)d_in[2];
  const float* rf_b     = (const float*)d_in[3];
  const float* pca_mean = (const float*)d_in[4];
  const float* pca_comp = (const float*)d_in[5];
  const float* W0 = (const float*)d_in[6],  *b0 = (const float*)d_in[7];
  const float* W1 = (const float*)d_in[8],  *b1 = (const float*)d_in[9];
  const float* W2 = (const float*)d_in[10], *b2 = (const float*)d_in[11];
  const float* W3 = (const float*)d_in[12], *b3 = (const float*)d_in[13];
  const float* nb0 = (const float*)d_in[14];
  const float* nb1 = (const float*)d_in[15];
  float* out = (float*)d_out;

  // workspace layout (1 MB = 1<<20 bytes); aliases noted
  char* ws = (char*)d_ws;
  float*    G    = (float*)   (ws);                  // 64MB (Grams; H1b alias)
  ushort_t* H1b  = (ushort_t*)(ws);                  // 32MB (dead before Grams)
  float*    H2   = (float*)   (ws + (64ull  << 20)); // 16MB
  ushort_t* H2b  = (ushort_t*)(ws + (80ull  << 20)); //  8MB
  float*    PEN  = (float*)   (ws + (88ull  << 20)); // 16MB
  ushort_t* PENb = (ushort_t*)(ws + (104ull << 20)); //  8MB
  ushort_t* Xb   = (ushort_t*)(ws + (112ull << 20)); //  8MB
  ushort_t* RWb  = (ushort_t*)(ws + (120ull << 20)); //  8MB (A0b alias)
  ushort_t* A0b  = RWb;
  ushort_t* PCb  = (ushort_t*)(ws + (128ull << 20)); //  8MB (A1b alias)
  ushort_t* A1b  = PCb;
  ushort_t* W0t  = (ushort_t*)(ws + (136ull << 20)); //  2MB
  ushort_t* W1t  = (ushort_t*)(ws + (138ull << 20)); //  2MB
  ushort_t* W2t  = (ushort_t*)(ws + (140ull << 20)); //  2MB
  float*    MB   = (float*)   (ws + (142ull << 20)); //  4KB
  float*    SQ   = (float*)   (ws + (142ull << 20) + 4096ull); // 16KB

  const dim3 blk(256);

  // --- precision staging: fp32 -> bf16 copies of all GEMM operands ---------
  cvt_bf16_kernel<<<dim3(N_ROWS * D_IN / 2048), blk, 0, stream>>>(X, Xb,
                                                                  N_ROWS * D_IN);
  cvt_bf16_kernel<<<dim3(RF_DIM * D_IN / 2048), blk, 0, stream>>>(rf_w, RWb,
                                                                  RF_DIM * D_IN);
  cvt_bf16_kernel<<<dim3(PC_DIM * RF_DIM / 2048), blk, 0, stream>>>(
      pca_comp, PCb, PC_DIM * RF_DIM);
  cvtT_bf16_kernel<<<dim3(PC_DIM * PC_DIM / 256), blk, 0, stream>>>(W0, W0t,
                                                                    PC_DIM, PC_DIM);
  cvtT_bf16_kernel<<<dim3(PC_DIM * PC_DIM / 256), blk, 0, stream>>>(W1, W1t,
                                                                    PC_DIM, PC_DIM);
  cvtT_bf16_kernel<<<dim3(PC_DIM * PC_DIM / 256), blk, 0, stream>>>(W2, W2t,
                                                                    PC_DIM, PC_DIM);
  // mb[p] = pca_mean . pca_comp[p,:]  (folds mean-subtract into S2 epilogue)
  rowdot_kernel<<<dim3(PC_DIM / 8), blk, 0, stream>>>(pca_mean, pca_comp, MB,
                                                      PC_DIM, RF_DIM);

  // --- trunk ---------------------------------------------------------------
  // H1b = bf16(relu(X @ rf_w^T + rf_b))
  gemm_bf16_wmma<0, 0, 1><<<dim3(RF_DIM / 128, N_ROWS / 256), blk, 0, stream>>>(
      Xb, RWb, rf_b, nullptr, nullptr, H1b, N_ROWS, RF_DIM, D_IN);
  // H2 = clip(H1 @ pca_comp^T - mb), fp32 + bf16 mirror
  gemm_bf16_wmma<1, 1, 1><<<dim3(PC_DIM / 128, N_ROWS / 256), blk, 0, stream>>>(
      H1b, PCb, MB, nullptr, H2, H2b, N_ROWS, PC_DIM, RF_DIM);
  // A0b = bf16(relu(H2 @ W0 + b0))
  gemm_bf16_wmma<0, 0, 1><<<dim3(PC_DIM / 128, N_ROWS / 256), blk, 0, stream>>>(
      H2b, W0t, b0, nullptr, nullptr, A0b, N_ROWS, PC_DIM, PC_DIM);
  // A1b = bf16(relu(A0 @ W1 + b1 + H2))
  gemm_bf16_wmma<2, 0, 1><<<dim3(PC_DIM / 128, N_ROWS / 256), blk, 0, stream>>>(
      A0b, W1t, b1, H2, nullptr, A1b, N_ROWS, PC_DIM, PC_DIM);
  // PEN = relu(A1 @ W2 + b2), fp32 + bf16 mirror
  gemm_bf16_wmma<0, 1, 1><<<dim3(PC_DIM / 128, N_ROWS / 256), blk, 0, stream>>>(
      A1b, W2t, b2, nullptr, PEN, PENb, N_ROWS, PC_DIM, PC_DIM);
  // out = PEN @ W3 + b3
  logits_kernel<<<dim3(N_ROWS), dim3(320), 0, stream>>>(PEN, W3, b3, out,
                                                        PC_DIM, NC_CLS);

  // --- LOO 1-NN bias 0: raw X ---------------------------------------------
  sqnorm_kernel<<<dim3(N_ROWS / 8), blk, 0, stream>>>(X, SQ, N_ROWS, D_IN);
  gemm_bf16_wmma<3, 1, 0><<<dim3(N_ROWS / 128, N_ROWS / 256), blk, 0, stream>>>(
      Xb, Xb, nullptr, nullptr, G, nullptr, N_ROWS, N_ROWS, D_IN);
  argmin_bias_kernel<<<dim3(N_ROWS), blk, 0, stream>>>(G, SQ, y, nb0, out,
                                                       N_ROWS, NC_CLS);
  // --- LOO 1-NN bias 1: penultimate activations ----------------------------
  sqnorm_kernel<<<dim3(N_ROWS / 8), blk, 0, stream>>>(PEN, SQ, N_ROWS, PC_DIM);
  gemm_bf16_wmma<3, 1, 0><<<dim3(N_ROWS / 128, N_ROWS / 256), blk, 0, stream>>>(
      PENb, PENb, nullptr, nullptr, G, nullptr, N_ROWS, N_ROWS, PC_DIM);
  argmin_bias_kernel<<<dim3(N_ROWS), blk, 0, stream>>>(G, SQ, y, nb1, out,
                                                       N_ROWS, NC_CLS);
}
